// Attention_Module_25340307046568
// MI455X (gfx1250) — matrix operand order repaired and checked by reading, not compiled, not run on hardware
//
#include <hip/hip_runtime.h>
#include <hip/hip_bf16.h>

typedef __bf16 bf16_t;
typedef __attribute__((ext_vector_type(16))) __bf16 v16bf;
typedef __attribute__((ext_vector_type(8)))  float  v8f;

union FragU { uint4 u[2]; unsigned w[8]; v16bf v; };

constexpr int Bv = 8, Cc = 512, Nn = 4096, Ss = 64;

// f32 -> bf16, round-half-up (one v_add + shift)
__device__ __forceinline__ bf16_t f2bf(float f) {
  unsigned u = __builtin_bit_cast(unsigned, f) + 0x8000u;
  unsigned short h = (unsigned short)(u >> 16);
  return __builtin_bit_cast(bf16_t, h);
}

// two f32 -> packed bf16x2 (a = low half): 2x v_add + 1x v_perm_b32
__device__ __forceinline__ unsigned pk2bf(float a, float b) {
  unsigned ua = __builtin_bit_cast(unsigned, a) + 0x8000u;
  unsigned ub = __builtin_bit_cast(unsigned, b) + 0x8000u;
  // {ub,ua} bytes: 0-3 = ua, 4-7 = ub; take ua[3:2] then ub[3:2]
  return __builtin_amdgcn_perm(ub, ua, 0x07060302u);
}

__device__ __forceinline__ v8f wmma_bf16(v16bf a, v16bf b, v8f c) {
  return __builtin_amdgcn_wmma_f32_16x16x32_bf16(false, a, false, b, (short)0, c,
                                                 false, false);
}

__device__ __forceinline__ v8f vzero8() {
  v8f z;
#pragma unroll
  for (int e = 0; e < 8; ++e) z[e] = 0.0f;
  return z;
}

// ---------------------------------------------------------------------------
// Kernel 0: one-shot f32 -> bf16 weight conversion.
// ---------------------------------------------------------------------------
__global__ __launch_bounds__(256) void cvt_w_kernel(
    const float* __restrict__ src, bf16_t* __restrict__ dst, int npairs)
{
  int i = blockIdx.x * 256 + threadIdx.x;
  if (i < npairs) {
    float2 v = *(const float2*)(src + 2 * i);
    *(unsigned*)(dst + 2 * i) = pk2bf(v.x, v.y);
  }
}

// ---------------------------------------------------------------------------
// Kernel 1: q/k/v projections.  q,k stored transposed [B][N][S] bf16; v [B][S][N].
// 384 threads = 12 waves = (4 S-row-tiles) x (q,k,v). 128 N-cols per block.
// ---------------------------------------------------------------------------
__global__ __launch_bounds__(384) void qkv_proj_kernel(
    const float* __restrict__ x,
    const bf16_t* __restrict__ wqb, const float* __restrict__ bq,
    const bf16_t* __restrict__ wkb, const float* __restrict__ bk,
    const bf16_t* __restrict__ wvb, const float* __restrict__ bv,
    bf16_t* __restrict__ qT, bf16_t* __restrict__ kT, bf16_t* __restrict__ vS)
{
  constexpr int TN = 128;   // n columns per block
  constexpr int KC = 32;    // channels per K step
  constexpr int LROW = 40;  // padded LDS row (bf16): 80B rows, 16B aligned
  __shared__ __align__(16) bf16_t xT[TN * LROW];   // [n][kk]

  const int b    = blockIdx.y;
  const int n0   = blockIdx.x * TN;
  const int t    = threadIdx.x;
  const int wave = t >> 5, lane = t & 31;
  const int which = wave % 3;
  const int mt    = wave / 3;
  const int half  = lane >> 4, l16 = lane & 15;

  const bf16_t* wb = (which == 0) ? wqb : (which == 1) ? wkb : wvb;
  const float*  bi = (which == 0) ? bq : (which == 1) ? bk : bv;

  v8f acc[8];
#pragma unroll
  for (int i = 0; i < 8; ++i) acc[i] = vzero8();

  const float* xb = x + (size_t)b * Cc * Nn + n0;
  const int srow  = mt * 16 + l16;

  for (int c0 = 0; c0 < Cc; c0 += KC) {
    __syncthreads();
    for (int i = t; i < (KC / 2) * TN; i += 384) {   // stage x tile (paired kk)
      int kp = i >> 7, nl = i & 127;
      float e0 = xb[(size_t)(c0 + 2 * kp) * Nn + nl];
      float e1 = xb[(size_t)(c0 + 2 * kp + 1) * Nn + nl];
      *(unsigned*)&xT[nl * LROW + 2 * kp] = pk2bf(e0, e1);
    }
    __syncthreads();

    FragU A;                                         // bf16 weights (global b128)
    const bf16_t* wp = wb + (size_t)srow * Cc + c0;
    A.u[0] = *(const uint4*)(wp + 16 * half);
    A.u[1] = *(const uint4*)(wp + 16 * half + 8);

    FragU Bf[8];                                     // preload all B fragments
#pragma unroll
    for (int nt = 0; nt < 8; ++nt) {
      const bf16_t* bp = &xT[(nt * 16 + l16) * LROW + 16 * half];
      Bf[nt].u[0] = *(const uint4*)(bp);
      Bf[nt].u[1] = *(const uint4*)(bp + 8);
    }
#pragma unroll
    for (int nt = 0; nt < 8; ++nt)
      acc[nt] = wmma_bf16(A.v, Bf[nt].v, acc[nt]);
  }

  if (which == 2) {
#pragma unroll
    for (int j = 0; j < 8; ++j) {
      const int sg = mt * 16 + j + 8 * half;
      const float bias = bi[sg];
#pragma unroll
      for (int nt = 0; nt < 8; ++nt) {
        const int ng = n0 + nt * 16 + l16;
        vS[((size_t)b * Ss + sg) * Nn + ng] = f2bf(acc[nt][j] + bias);
      }
    }
  } else {
    bf16_t* dst = (which == 0) ? qT : kT;
#pragma unroll
    for (int nt = 0; nt < 8; ++nt) {
      const int ng = n0 + nt * 16 + l16;
      bf16_t* rp = dst + ((size_t)b * Nn + ng) * Ss;
#pragma unroll
      for (int j = 0; j < 8; j += 2) {               // sg consecutive -> b32 store
        const int sg = mt * 16 + j + 8 * half;
        *(unsigned*)(rp + sg) =
            pk2bf(acc[nt][j] + bi[sg], acc[nt][j + 1] + bi[sg + 1]);
      }
    }
  }
}

// ---------------------------------------------------------------------------
// Kernel 2: att[b,n,m] = sum_s k[b,s,n] * q[b,s,m]  (f32 logits to d_out).
// The 8KB q tile is shared by all 4 waves -> async-DMA it into LDS once.
// ---------------------------------------------------------------------------
__global__ __launch_bounds__(128) void att_gemm_kernel(
    const bf16_t* __restrict__ qT, const bf16_t* __restrict__ kT,
    float* __restrict__ att)
{
  __shared__ __align__(16) bf16_t qTile[64 * Ss];   // 8 KB

  const int b  = blockIdx.z;
  const int nB = blockIdx.y * 64;
  const int mB = blockIdx.x * 64;
  const int t  = threadIdx.x;
  const int wave = t >> 5, lane = t & 31;
  const int half = lane >> 4, l16 = lane & 15;

  const bf16_t* kb = kT + ((size_t)b * Nn + nB) * Ss;
  const bf16_t* qb = qT + ((size_t)b * Nn + mB) * Ss;

  {   // async copy: 64 contiguous bytes per thread (tile is fully contiguous)
    unsigned lds = (unsigned)(size_t)&qTile[t * 32];
    unsigned long long ga = (unsigned long long)(size_t)(qb + t * 32);
    asm volatile(
        "global_load_async_to_lds_b128 %0, %1, off\n\t"
        "global_load_async_to_lds_b128 %0, %1, off offset:16\n\t"
        "global_load_async_to_lds_b128 %0, %1, off offset:32\n\t"
        "global_load_async_to_lds_b128 %0, %1, off offset:48"
        :: "v"(lds), "v"(ga) : "memory");
    asm volatile("s_wait_asynccnt 0x0" ::: "memory");
  }
  __syncthreads();

  v8f acc[4];
#pragma unroll
  for (int i = 0; i < 4; ++i) acc[i] = vzero8();

#pragma unroll
  for (int ks = 0; ks < Ss; ks += 32) {
    FragU A;                                         // kT rows (global b128)
    const bf16_t* ap = kb + (size_t)(wave * 16 + l16) * Ss + ks;
    A.u[0] = *(const uint4*)(ap + 16 * half);
    A.u[1] = *(const uint4*)(ap + 16 * half + 8);

    FragU Bf[4];                                     // q tile (LDS b128)
#pragma unroll
    for (int mt = 0; mt < 4; ++mt) {
      const bf16_t* bp = &qTile[(mt * 16 + l16) * Ss + ks + 16 * half];
      Bf[mt].u[0] = *(const uint4*)(bp);
      Bf[mt].u[1] = *(const uint4*)(bp + 8);
    }
#pragma unroll
    for (int mt = 0; mt < 4; ++mt)
      acc[mt] = wmma_bf16(A.v, Bf[mt].v, acc[mt]);
  }

  float* ab = att + ((size_t)b * Nn + nB) * Nn + mB;
#pragma unroll
  for (int mt = 0; mt < 4; ++mt)
#pragma unroll
    for (int j = 0; j < 8; ++j)
      ab[(size_t)(wave * 16 + j + 8 * half) * Nn + mt * 16 + l16] = acc[mt][j];
}

// ---------------------------------------------------------------------------
// Kernel 3: row softmax over m (4096 per row), in place on f32 att.
// ---------------------------------------------------------------------------
__global__ __launch_bounds__(256) void softmax_kernel(float* __restrict__ att)
{
  float* p = att + (size_t)blockIdx.x * Nn;
  const int t = threadIdx.x;
  __shared__ float red[8];

  float4 vals[4];
  float m = -INFINITY;
#pragma unroll
  for (int i = 0; i < 4; ++i) {
    vals[i] = *(const float4*)(p + i * 1024 + t * 4);
    m = fmaxf(m, fmaxf(fmaxf(vals[i].x, vals[i].y), fmaxf(vals[i].z, vals[i].w)));
  }
#pragma unroll
  for (int o = 16; o > 0; o >>= 1) m = fmaxf(m, __shfl_xor(m, o, 32));
  if ((t & 31) == 0) red[t >> 5] = m;
  __syncthreads();
  float bm = red[0];
#pragma unroll
  for (int i = 1; i < 8; ++i) bm = fmaxf(bm, red[i]);
  __syncthreads();

  float s = 0.0f;
#pragma unroll
  for (int i = 0; i < 4; ++i) {
    vals[i].x = __expf(vals[i].x - bm);
    vals[i].y = __expf(vals[i].y - bm);
    vals[i].z = __expf(vals[i].z - bm);
    vals[i].w = __expf(vals[i].w - bm);
    s += vals[i].x + vals[i].y + vals[i].z + vals[i].w;
  }
#pragma unroll
  for (int o = 16; o > 0; o >>= 1) s += __shfl_xor(s, o, 32);
  if ((t & 31) == 0) red[t >> 5] = s;
  __syncthreads();
  float total = 0.0f;
#pragma unroll
  for (int i = 0; i < 8; ++i) total += red[i];
  const float inv = 1.0f / total;

#pragma unroll
  for (int i = 0; i < 4; ++i) {
    vals[i].x *= inv; vals[i].y *= inv; vals[i].z *= inv; vals[i].w *= inv;
    *(float4*)(p + i * 1024 + t * 4) = vals[i];
  }
}

// ---------------------------------------------------------------------------
// Kernel 4: v2[b,s,m] = sum_n v[b,s,n] * att[b,n,m], K = 4096.
// att (f32) staged+converted through LDS; output transposed [B][N][S] bf16.
// ---------------------------------------------------------------------------
__global__ __launch_bounds__(128) void av_gemm_kernel(
    const bf16_t* __restrict__ vS, const float* __restrict__ att,
    bf16_t* __restrict__ v2T)
{
  constexpr int LROW = 40;
  __shared__ __align__(16) bf16_t aT[64 * LROW];    // [m][kk]

  const int b  = blockIdx.y;
  const int mB = blockIdx.x * 64;
  const int wave = threadIdx.x >> 5, lane = threadIdx.x & 31;
  const int half = lane >> 4, l16 = lane & 15;

  const float*  ab = att + (size_t)b * Nn * Nn + mB;
  const bf16_t* vb = vS + (size_t)b * Ss * Nn;

  v8f acc[4];
#pragma unroll
  for (int i = 0; i < 4; ++i) acc[i] = vzero8();

  for (int kn = 0; kn < Nn; kn += 32) {
    __syncthreads();
    for (int i = threadIdx.x; i < 16 * 64; i += 128) {  // paired kk staging
      int kp = i >> 6, ml = i & 63;
      float e0 = ab[(size_t)(kn + 2 * kp) * Nn + ml];
      float e1 = ab[(size_t)(kn + 2 * kp + 1) * Nn + ml];
      *(unsigned*)&aT[ml * LROW + 2 * kp] = pk2bf(e0, e1);
    }
    if (kn + 32 < Nn)
      __builtin_prefetch(&ab[(size_t)(kn + 32) * Nn + (threadIdx.x << 4)], 0, 1);
    __syncthreads();

    FragU A;                                        // v rows (global b128)
    const bf16_t* ap = vb + (size_t)(wave * 16 + l16) * Nn + kn;
    A.u[0] = *(const uint4*)(ap + 16 * half);
    A.u[1] = *(const uint4*)(ap + 16 * half + 8);

    FragU Bf[4];
#pragma unroll
    for (int mt = 0; mt < 4; ++mt) {
      const bf16_t* bp = &aT[(mt * 16 + l16) * LROW + 16 * half];
      Bf[mt].u[0] = *(const uint4*)(bp);
      Bf[mt].u[1] = *(const uint4*)(bp + 8);
    }
#pragma unroll
    for (int mt = 0; mt < 4; ++mt)
      acc[mt] = wmma_bf16(A.v, Bf[mt].v, acc[mt]);
  }

#pragma unroll
  for (int mt = 0; mt < 4; ++mt) {
    const int mg = mB + mt * 16 + l16;
    bf16_t* rp = v2T + ((size_t)b * Nn + mg) * Ss;
#pragma unroll
    for (int j = 0; j < 8; j += 2) {                // paired b32 stores
      const int sg = wave * 16 + j + 8 * half;
      *(unsigned*)(rp + sg) = pk2bf(acc[mt][j], acc[mt][j + 1]);
    }
  }
}

// ---------------------------------------------------------------------------
// Kernel 5: out = coef * (wa @ v2 + ba) + x.  K = S = 64.
// The 8KB v2 tile is shared by all 8 waves -> async-DMA it into LDS once.
// ---------------------------------------------------------------------------
__global__ __launch_bounds__(256) void out_proj_kernel(
    const float* __restrict__ x, const bf16_t* __restrict__ wab,
    const float* __restrict__ ba, const float* __restrict__ coef,
    const bf16_t* __restrict__ v2T, float* __restrict__ out)
{
  __shared__ __align__(16) bf16_t v2Tile[64 * Ss];  // 8 KB

  const int b  = blockIdx.z;
  const int c0 = blockIdx.y * 128;
  const int mB = blockIdx.x * 64;
  const int t  = threadIdx.x;
  const int wave = t >> 5, lane = t & 31;
  const int half = lane >> 4, l16 = lane & 15;

  const int crow = c0 + wave * 16 + l16;
  const bf16_t* v2b = v2T + ((size_t)b * Nn + mB) * Ss;

  {   // async copy: 32 contiguous bytes per thread (256 threads, 8 KB tile)
    unsigned lds = (unsigned)(size_t)&v2Tile[t * 16];
    unsigned long long ga = (unsigned long long)(size_t)(v2b + t * 16);
    asm volatile(
        "global_load_async_to_lds_b128 %0, %1, off\n\t"
        "global_load_async_to_lds_b128 %0, %1, off offset:16"
        :: "v"(lds), "v"(ga) : "memory");
    asm volatile("s_wait_asynccnt 0x0" ::: "memory");
  }
  __syncthreads();

  v8f acc[4];
#pragma unroll
  for (int i = 0; i < 4; ++i) acc[i] = vzero8();

#pragma unroll
  for (int ks = 0; ks < Ss; ks += 32) {
    FragU A;                                        // bf16 wa rows (global b128)
    const bf16_t* wp = wab + (size_t)crow * Ss + ks;
    A.u[0] = *(const uint4*)(wp + 16 * half);
    A.u[1] = *(const uint4*)(wp + 16 * half + 8);

    FragU Bf[4];                                    // v2 tile (LDS b128)
#pragma unroll
    for (int mt = 0; mt < 4; ++mt) {
      const bf16_t* bp = &v2Tile[(mt * 16 + l16) * Ss + ks + 16 * half];
      Bf[mt].u[0] = *(const uint4*)(bp);
      Bf[mt].u[1] = *(const uint4*)(bp + 8);
    }
#pragma unroll
    for (int mt = 0; mt < 4; ++mt)
      acc[mt] = wmma_bf16(A.v, Bf[mt].v, acc[mt]);
  }

  const float cf = coef[0];
#pragma unroll
  for (int mt = 0; mt < 4; ++mt)
#pragma unroll
    for (int j = 0; j < 8; ++j) {
      const int cg = c0 + wave * 16 + j + 8 * half;
      const int mg = mB + mt * 16 + l16;
      const size_t idx = ((size_t)b * Cc + cg) * Nn + mg;
      out[idx] = cf * (acc[mt][j] + ba[cg]) + x[idx];
    }
}

// ---------------------------------------------------------------------------
extern "C" void kernel_launch(void* const* d_in, const int* in_sizes, int n_in,
                              void* d_out, int out_size, void* d_ws, size_t ws_size,
                              hipStream_t stream) {
  const float* x    = (const float*)d_in[0];
  const float* wq   = (const float*)d_in[1];
  const float* bq   = (const float*)d_in[2];
  const float* wk   = (const float*)d_in[3];
  const float* bk   = (const float*)d_in[4];
  const float* wv   = (const float*)d_in[5];
  const float* bv   = (const float*)d_in[6];
  const float* wa   = (const float*)d_in[7];
  const float* ba   = (const float*)d_in[8];
  const float* coef = (const float*)d_in[9];

  float* out = (float*)d_out;
  float* att = out + (size_t)Bv * Cc * Nn;

  bf16_t* qT  = (bf16_t*)d_ws;                 // [B][N][S]
  bf16_t* kT  = qT + (size_t)Bv * Nn * Ss;     // [B][N][S]
  bf16_t* vS  = kT + (size_t)Bv * Nn * Ss;     // [B][S][N]
  bf16_t* v2T = vS + (size_t)Bv * Nn * Ss;     // [B][N][S]
  bf16_t* wqb = v2T + (size_t)Bv * Nn * Ss;    // [S][C]
  bf16_t* wkb = wqb + (size_t)Ss * Cc;
  bf16_t* wvb = wkb + (size_t)Ss * Cc;
  bf16_t* wab = wvb + (size_t)Ss * Cc;         // [C][S]   (~16.25 MB total)

  const int wp = (Ss * Cc) / 2;                // 16384 pairs per weight matrix
  cvt_w_kernel<<<dim3((wp + 255) / 256), 256, 0, stream>>>(wq, wqb, wp);
  cvt_w_kernel<<<dim3((wp + 255) / 256), 256, 0, stream>>>(wk, wkb, wp);
  cvt_w_kernel<<<dim3((wp + 255) / 256), 256, 0, stream>>>(wv, wvb, wp);
  cvt_w_kernel<<<dim3((wp + 255) / 256), 256, 0, stream>>>(wa, wab, wp);

  qkv_proj_kernel<<<dim3(Nn / 128, Bv), 384, 0, stream>>>(
      x, wqb, bq, wkb, bk, wvb, bv, qT, kT, vS);
  att_gemm_kernel<<<dim3(Nn / 64, Nn / 64, Bv), 128, 0, stream>>>(qT, kT, att);
  softmax_kernel<<<dim3(Bv * Nn), 256, 0, stream>>>(att);
  av_gemm_kernel<<<dim3(Nn / 64, Bv), 128, 0, stream>>>(vS, att, v2T);
  out_proj_kernel<<<dim3(Nn / 64, Cc / 128, Bv), 256, 0, stream>>>(
      x, wab, ba, coef, v2T, out);
}
